// Extraction_Network_69930657513510
// MI455X (gfx1250) — matrix operand order repaired and checked by reading, not compiled, not run on hardware
//
#include <hip/hip_runtime.h>
#include <hip/hip_bf16.h>

// ---------------------------------------------------------------------------
// MoE extraction network for MI455X (gfx1250), wave32 + v_wmma_f32_16x16x32_bf16
// 32 batch rows per workgroup, dual M-tiles per weight-fragment load (2x B reuse)
// ---------------------------------------------------------------------------

typedef __attribute__((ext_vector_type(16))) __bf16       v16bf;
typedef __attribute__((ext_vector_type(4)))  __bf16       bf16x4;
typedef __attribute__((ext_vector_type(8)))  float        v8f;
typedef __attribute__((ext_vector_type(4)))  float        f32x4;
typedef __attribute__((ext_vector_type(4)))  unsigned int u32x4;

union V16 { v16bf v; u32x4 q[2]; };

#define B_SZ 16384
#define F_SZ 1024
#define H_SZ 64
#define D_SZ 256

// Packed-weight workspace offsets (bf16 elements).
// B-fragment tile = 32(K)x16(N), stored as [lane(32)][elem(16)] = 1KB.
#define PW1_GSZ 262144           // per group: 16 ntiles * 32 ktiles * 512
#define PWG_OFF 786432           // 3*PW1_GSZ
#define PWG_GSZ 16384            // per group: 32 ktiles * 512
#define PW2_OFF 835584           // PWG_OFF + 3*PWG_GSZ
#define PW2_GSZ 65536            // per group: 4e * 16nt * 2kt * 512
// total = 1,032,192 bf16 = ~2.06 MB of d_ws

// LDS layout (bytes), 32 rows per WG
#define XLS_GSZ  (32 * 1024)                 // bf16 elems per group
#define HLS_GSZ  (32 * 256)                  // bf16 elems per group
#define XLS_BYTES (3 * XLS_GSZ * 2)          // 196608
#define HLS_BYTES (3 * HLS_GSZ * 2)          // 49152
#define GLS_BYTES (3 * 32 * 16 * 4)          // 6144
#define LDS_BYTES (XLS_BYTES + HLS_BYTES + GLS_BYTES)   // 251904

// ---------------------------------------------------------------------------
// Prep: convert fp32 weights -> bf16, pre-swizzled into WMMA B-fragment order.
// Per-lane 32B = 16 consecutive K values of one column n = lane%16;
// lanes 16-31 hold K+16 (contiguous k order, matched by the A loads).
// ---------------------------------------------------------------------------
__global__ void pack_weights(const float* __restrict__ W1A, const float* __restrict__ W1S,
                             const float* __restrict__ W1B,
                             const float* __restrict__ WgA, const float* __restrict__ WgS,
                             const float* __restrict__ WgB,
                             const float* __restrict__ W2A, const float* __restrict__ W2S,
                             const float* __restrict__ W2B,
                             __bf16* __restrict__ pk) {
  int tid = blockIdx.x * blockDim.x + threadIdx.x;
  if (tid >= 2016 * 32) return;
  int tile = tid >> 5, lane = tid & 31;
  int hh = lane >> 4, n16 = lane & 15;

  if (tile < 1536) {                       // layer-1 expert weights [E,F,H]
    int g = tile / 512, r = tile % 512;
    int t = r >> 5, kt = r & 31;
    const float* W1 = (g == 0) ? W1A : ((g == 1) ? W1S : W1B);
    int e = t >> 2;
    int h = (t & 3) * 16 + n16;
    int fbase = kt * 32 + hh * 16;
    __bf16* dst = pk + (size_t)g * PW1_GSZ + ((size_t)(t * 32 + kt) * 32 + lane) * 16;
#pragma unroll
    for (int i = 0; i < 16; ++i)
      dst[i] = (__bf16)W1[((size_t)e * F_SZ + (fbase + i)) * H_SZ + h];
  } else if (tile < 1632) {                // gate weights [F, 8|12], pad N to 16
    int u = tile - 1536;
    int g = u >> 5, kt = u & 31;
    const float* Wg = (g == 0) ? WgA : ((g == 1) ? WgS : WgB);
    int ncols = (g == 1) ? 12 : 8;
    int fbase = kt * 32 + hh * 16;
    __bf16* dst = pk + PWG_OFF + (size_t)g * PWG_GSZ + ((size_t)(kt * 32) + lane) * 16;
#pragma unroll
    for (int i = 0; i < 16; ++i) {
      float v = (n16 < ncols) ? Wg[(size_t)(fbase + i) * ncols + n16] : 0.0f;
      dst[i] = (__bf16)v;
    }
  } else {                                 // layer-2 expert weights [E,H,D]
    int u = tile - 1632;
    int g = u >> 7, r = u & 127;
    int e = r >> 5, r2 = r & 31;
    int nt = r2 >> 1, kt = r2 & 1;
    const float* W2 = (g == 0) ? W2A : ((g == 1) ? W2S : W2B);
    int d = nt * 16 + n16;
    int kbase = kt * 32 + hh * 16;
    __bf16* dst = pk + PW2_OFF + (size_t)g * PW2_GSZ +
                  ((size_t)(e * 32 + nt * 2 + kt) * 32 + lane) * 16;
#pragma unroll
    for (int i = 0; i < 16; ++i)
      dst[i] = (__bf16)W2[((size_t)e * H_SZ + (kbase + i)) * D_SZ + d];
  }
}

// ---------------------------------------------------------------------------
// Fused MoE kernel: one workgroup = 32 batch rows (two 16-row M-tiles),
// 512 threads = 16 waves. Each weight B-fragment feeds two WMMAs.
// ---------------------------------------------------------------------------
__global__ __launch_bounds__(512)
void moe_fused(const float* __restrict__ xA, const float* __restrict__ xS,
               const float* __restrict__ xB,
               const float* __restrict__ b1A, const float* __restrict__ b1S,
               const float* __restrict__ b1B,
               const float* __restrict__ b2A, const float* __restrict__ b2S,
               const float* __restrict__ b2B,
               const float* __restrict__ bgA, const float* __restrict__ bgS,
               const float* __restrict__ bgB,
               const __bf16* __restrict__ pk, float* __restrict__ out) {
  extern __shared__ char smem[];
  __bf16* xls = (__bf16*)smem;                          // [3][32][1024]
  __bf16* hls = (__bf16*)(smem + XLS_BYTES);            // [3][32][256]
  float*  gls = (float*)(smem + XLS_BYTES + HLS_BYTES); // [3][32][16]

  const int tid  = threadIdx.x;
  const int lane = tid & 31;
  const int wave = tid >> 5;
  const int hh   = lane >> 4;     // wave half (0/1)
  const int n16  = lane & 15;
  const int row0 = blockIdx.x * 32;

  // ---- Phase 0: stage x tiles, fp32 -> bf16 in LDS (nontemporal reads) -----
  {
    const float* xs[3] = {xA, xS, xB};
    for (int g = 0; g < 3; ++g) {
      const f32x4* src = (const f32x4*)(xs[g] + (size_t)row0 * F_SZ);
      for (int i = tid; i < 32 * 256; i += 512) {   // 8192 float4 per group
        f32x4 v = __builtin_nontemporal_load(&src[i]);
        bf16x4 p = {(__bf16)v.x, (__bf16)v.y, (__bf16)v.z, (__bf16)v.w};
        *(bf16x4*)(xls + (size_t)g * XLS_GSZ + (size_t)i * 4) = p;
      }
    }
  }
  __syncthreads();

  // ---- Phase 1: layer-1 GEMM (48 expert col-tiles) + gate logits (3 tiles) -
  for (int t = wave; t < 51; t += 16) {
    int g;
    const __bf16* bbase;
    if (t < 48) { g = t >> 4; int tt = t & 15;
      bbase = pk + (size_t)g * PW1_GSZ + (size_t)tt * 32 * 512;
    } else { g = t - 48;
      bbase = pk + PWG_OFF + (size_t)g * PWG_GSZ;
    }
    const __bf16* arow0 = xls + (size_t)g * XLS_GSZ + (size_t)n16 * 1024;  // rows 0-15
    const __bf16* arow1 = arow0 + 16 * 1024;                               // rows 16-31
    v8f c0 = {}, c1 = {};
    for (int kt = 0; kt < 32; ++kt) {
      V16 a0, a1, b;
      int k0 = kt * 32 + hh * 16;                    // one contiguous 16-elem K run
      const __bf16* bp = bbase + ((size_t)kt * 32 + lane) * 16;
      b.q[0]  = ((const u32x4*)bp)[0];
      b.q[1]  = ((const u32x4*)bp)[1];
      a0.q[0] = *(const u32x4*)(arow0 + k0);
      a0.q[1] = *(const u32x4*)(arow0 + k0 + 8);
      a1.q[0] = *(const u32x4*)(arow1 + k0);
      a1.q[1] = *(const u32x4*)(arow1 + k0 + 8);
      c0 = __builtin_amdgcn_wmma_f32_16x16x32_bf16(false, a0.v, false, b.v,
                                                   (short)0, c0, false, false);
      c1 = __builtin_amdgcn_wmma_f32_16x16x32_bf16(false, a1.v, false, b.v,
                                                   (short)0, c1, false, false);
    }
    if (t < 48) {                                    // bias + relu -> h in LDS
      int tt = t & 15;
      const float* b1 = (g == 0) ? b1A : ((g == 1) ? b1S : b1B);
      int e = tt >> 2;
      float bias = b1[e * H_SZ + (tt & 3) * 16 + n16];
      __bf16* hdst = hls + (size_t)g * HLS_GSZ + (tt * 16 + n16);
#pragma unroll
      for (int v = 0; v < 8; ++v) {
        float v0 = c0[v] + bias; v0 = v0 > 0.f ? v0 : 0.f;
        float v1 = c1[v] + bias; v1 = v1 > 0.f ? v1 : 0.f;
        hdst[(size_t)(v + hh * 8) * 256]      = (__bf16)v0;
        hdst[(size_t)(16 + v + hh * 8) * 256] = (__bf16)v1;
      }
    } else {                                         // gate logits -> LDS
      const float* bg = (g == 0) ? bgA : ((g == 1) ? bgS : bgB);
      int nc = (g == 1) ? 12 : 8;
      float bias = (n16 < nc) ? bg[n16] : 0.0f;
      float* gdst = gls + g * 512 + n16;
#pragma unroll
      for (int v = 0; v < 8; ++v) {
        gdst[(v + hh * 8) * 16]      = c0[v] + bias;
        gdst[(16 + v + hh * 8) * 16] = c1[v] + bias;
      }
    }
  }
  __syncthreads();

  // ---- Phase 2: per-row softmax of gates (96 rows) -------------------------
  if (tid < 96) {
    int g = tid >> 5, m = tid & 31;
    int nc = (g == 1) ? 12 : 8;
    float* row = gls + g * 512 + m * 16;
    float mx = row[0];
    for (int j = 1; j < nc; ++j) mx = fmaxf(mx, row[j]);
    float s = 0.f;
    for (int j = 0; j < nc; ++j) s += __expf(row[j] - mx);
    float inv = 1.f / s;
    for (int j = 0; j < nc; ++j) row[j] = __expf(row[j] - mx) * inv;
  }
  __syncthreads();

  // ---- Phase 3: layer-2 GEMM + gate-weighted mix; wave = one 16-col D-tile -
  {
    const int dt = wave;                             // 0..15, D-tile
    v8f accA0 = {}, accS0 = {}, accB0 = {};
    v8f accA1 = {}, accS1 = {}, accB1 = {};
#pragma unroll
    for (int slot = 0; slot < 12; ++slot) {          // EA0..3, ES0..3, EB0..3
      int g = slot >> 2, e = slot & 3;
      const __bf16* arow0 = hls + (size_t)g * HLS_GSZ + (size_t)n16 * 256 + e * 64;
      const __bf16* arow1 = arow0 + 16 * 256;
      v8f c0 = {}, c1 = {};
#pragma unroll
      for (int kt = 0; kt < 2; ++kt) {               // K = 64
        V16 a0, a1, b;
        int k0 = kt * 32 + hh * 16;
        const __bf16* bp = pk + PW2_OFF + (size_t)g * PW2_GSZ +
                           ((size_t)(e * 32 + dt * 2 + kt) * 32 + lane) * 16;
        b.q[0]  = ((const u32x4*)bp)[0];
        b.q[1]  = ((const u32x4*)bp)[1];
        a0.q[0] = *(const u32x4*)(arow0 + k0);
        a0.q[1] = *(const u32x4*)(arow0 + k0 + 8);
        a1.q[0] = *(const u32x4*)(arow1 + k0);
        a1.q[1] = *(const u32x4*)(arow1 + k0 + 8);
        c0 = __builtin_amdgcn_wmma_f32_16x16x32_bf16(false, a0.v, false, b.v,
                                                     (short)0, c0, false, false);
        c1 = __builtin_amdgcn_wmma_f32_16x16x32_bf16(false, a1.v, false, b.v,
                                                     (short)0, c1, false, false);
      }
      const float* b2 = (g == 0) ? b2A : ((g == 1) ? b2S : b2B);
      float bias = b2[e * D_SZ + dt * 16 + n16];
      const float* gA_ = gls;
      const float* gS_ = gls + 512;
      const float* gB_ = gls + 1024;
#pragma unroll
      for (int v = 0; v < 8; ++v) {
        int m0 = v + hh * 8;
        int m1 = m0 + 16;
        float E0 = c0[v] + bias; E0 = E0 > 0.f ? E0 : 0.f;
        float E1 = c1[v] + bias; E1 = E1 > 0.f ? E1 : 0.f;
        accS0[v] += gS_[m0 * 16 + slot] * E0;        // gS: concat(EA,ES,EB)
        accS1[v] += gS_[m1 * 16 + slot] * E1;
        if (g == 0) {                                // gA: EA part
          accA0[v] += gA_[m0 * 16 + e] * E0;
          accA1[v] += gA_[m1 * 16 + e] * E1;
        } else if (g == 1) {                         // gA/gB: ES part
          accA0[v] += gA_[m0 * 16 + 4 + e] * E0;
          accA1[v] += gA_[m1 * 16 + 4 + e] * E1;
          accB0[v] += gB_[m0 * 16 + 4 + e] * E0;
          accB1[v] += gB_[m1 * 16 + 4 + e] * E1;
        } else {                                     // gB: EB part
          accB0[v] += gB_[m0 * 16 + e] * E0;
          accB1[v] += gB_[m1 * 16 + e] * E1;
        }
      }
    }
    float* outA = out;
    float* outS = out + (size_t)B_SZ * D_SZ;
    float* outB = out + (size_t)2 * B_SZ * D_SZ;
    int col = dt * 16 + n16;
#pragma unroll
    for (int v = 0; v < 8; ++v) {
      size_t i0 = (size_t)(row0 + v + hh * 8) * D_SZ + col;
      size_t i1 = i0 + (size_t)16 * D_SZ;
      __builtin_nontemporal_store(accA0[v], &outA[i0]);
      __builtin_nontemporal_store(accA1[v], &outA[i1]);
      __builtin_nontemporal_store(accS0[v], &outS[i0]);
      __builtin_nontemporal_store(accS1[v], &outS[i1]);
      __builtin_nontemporal_store(accB0[v], &outB[i0]);
      __builtin_nontemporal_store(accB1[v], &outB[i1]);
    }
  }
}

// ---------------------------------------------------------------------------
extern "C" void kernel_launch(void* const* d_in, const int* in_sizes, int n_in,
                              void* d_out, int out_size, void* d_ws, size_t ws_size,
                              hipStream_t stream) {
  // dict order: x_A,x_S,x_B, W1A,b1A,W2A,b2A, W1S,b1S,W2S,b2S, W1B,b1B,W2B,b2B,
  //             WgA,bgA, WgB,bgB, WgS,bgS
  const float* xA  = (const float*)d_in[0];
  const float* xS  = (const float*)d_in[1];
  const float* xB  = (const float*)d_in[2];
  const float* W1A = (const float*)d_in[3];
  const float* b1A = (const float*)d_in[4];
  const float* W2A = (const float*)d_in[5];
  const float* b2A = (const float*)d_in[6];
  const float* W1S = (const float*)d_in[7];
  const float* b1S = (const float*)d_in[8];
  const float* W2S = (const float*)d_in[9];
  const float* b2S = (const float*)d_in[10];
  const float* W1B = (const float*)d_in[11];
  const float* b1B = (const float*)d_in[12];
  const float* W2B = (const float*)d_in[13];
  const float* b2B = (const float*)d_in[14];
  const float* WgA = (const float*)d_in[15];
  const float* bgA = (const float*)d_in[16];
  const float* WgB = (const float*)d_in[17];
  const float* bgB = (const float*)d_in[18];
  const float* WgS = (const float*)d_in[19];
  const float* bgS = (const float*)d_in[20];

  __bf16* pk = (__bf16*)d_ws;   // needs ~2.06 MB of workspace

  // Prep: pack weights into WMMA B-fragment layout (bf16). 2016 tiles * 32 lanes.
  pack_weights<<<252, 256, 0, stream>>>(W1A, W1S, W1B, WgA, WgS, WgB,
                                        W2A, W2S, W2B, pk);

  (void)hipFuncSetAttribute((const void*)moe_fused,
                            hipFuncAttributeMaxDynamicSharedMemorySize,
                            (int)LDS_BYTES);

  moe_fused<<<B_SZ / 32, 512, LDS_BYTES, stream>>>(xA, xS, xB,
                                                   b1A, b1S, b1B,
                                                   b2A, b2S, b2B,
                                                   bgA, bgS, bgB,
                                                   pk, (float*)d_out);
}
